// TransformerMultiHead_38963943309632
// MI455X (gfx1250) — matrix operand order repaired and checked by reading, not compiled, not run on hardware
//
#include <hip/hip_runtime.h>
#include <cmath>

typedef __attribute__((ext_vector_type(16))) _Float16 v16h;
typedef __attribute__((ext_vector_type(8)))  _Float16 h8;
typedef __attribute__((ext_vector_type(2)))  _Float16 h2;
typedef __attribute__((ext_vector_type(8)))  float    v8f;
typedef __attribute__((ext_vector_type(4)))  int      v4i;

typedef __attribute__((address_space(1))) v4i gv4i;   // global int4
typedef __attribute__((address_space(3))) v4i lv4i;   // LDS int4

#define LAY   12
#define DIM   1024
#define NHEAD 16
#define HDIM  64
#define TSEQ  512
#define BATCH 8
#define ROWS  (BATCH*TSEQ)   /* 4096 */
#define FF    (4*DIM)        /* 4096 */
#define CLS   256
#define NOUT  4
#define LNEPS 1e-5f

// ---- CDNA5 async global->LDS copy (ASYNCcnt) with safe fallback ------------
#if __has_builtin(__builtin_amdgcn_global_load_async_to_lds_b128)
#define USE_ASYNC_LDS 1
#define ASYNC_B128(gp, lp)                                                     \
  __builtin_amdgcn_global_load_async_to_lds_b128(                              \
      (gv4i*)const_cast<_Float16*>(gp), (lv4i*)(lp), 0, 0)
#if __has_builtin(__builtin_amdgcn_s_wait_asynccnt)
#define WAIT_ASYNC() __builtin_amdgcn_s_wait_asynccnt(0)
#else
#define WAIT_ASYNC() asm volatile("s_wait_asynccnt 0" ::: "memory")
#endif
#else
#define USE_ASYNC_LDS 0
#define ASYNC_B128(gp, lp) (*(h8*)(lp) = *(const h8*)(gp))
#define WAIT_ASYNC() ((void)0)
#endif

// ---------------------------------------------------------------------------
// Embedding sum: x[i,:] = tok[idx[i]] + pos[t] + segm[st[i]] + tex[tt[i]]
// ---------------------------------------------------------------------------
__global__ void embed_kernel(const int* __restrict__ idx,
                             const int* __restrict__ st,
                             const int* __restrict__ tt,
                             const float* __restrict__ tok,
                             const float* __restrict__ pos,
                             const float* __restrict__ sem,
                             const float* __restrict__ tex,
                             float* __restrict__ x) {
  int i = blockIdx.x;                 // 0..ROWS-1
  int t = i & (TSEQ - 1);
  const float* tr = tok + (size_t)idx[i] * DIM;
  const float* pr = pos + (size_t)t      * DIM;
  const float* sr = sem + (size_t)st[i]  * DIM;
  const float* xr = tex + (size_t)tt[i]  * DIM;
  float* o = x + (size_t)i * DIM;
  for (int d = threadIdx.x; d < DIM; d += blockDim.x)
    o[d] = tr[d] + pr[d] + sr[d] + xr[d];
}

// ---------------------------------------------------------------------------
// Row LayerNorm over DIM, f32 in -> f16 out (feeds WMMA GEMM A operand)
// ---------------------------------------------------------------------------
__global__ __launch_bounds__(256)
void ln_kernel(const float* __restrict__ x, const float* __restrict__ w,
               const float* __restrict__ b, _Float16* __restrict__ outh) {
  __shared__ float red[256], red2[256];
  int row = blockIdx.x;
  const float* xr = x + (size_t)row * DIM;
  float s = 0.f, s2 = 0.f;
  for (int d = threadIdx.x; d < DIM; d += blockDim.x) {
    float v = xr[d]; s += v; s2 += v * v;
  }
  red[threadIdx.x] = s; red2[threadIdx.x] = s2;
  __syncthreads();
  for (int off = 128; off > 0; off >>= 1) {
    if (threadIdx.x < off) {
      red[threadIdx.x]  += red[threadIdx.x + off];
      red2[threadIdx.x] += red2[threadIdx.x + off];
    }
    __syncthreads();
  }
  float mu  = red[0] * (1.0f / DIM);
  float var = red2[0] * (1.0f / DIM) - mu * mu;
  float inv = rsqrtf(var + LNEPS);
  _Float16* o = outh + (size_t)row * DIM;
  for (int d = threadIdx.x; d < DIM; d += blockDim.x)
    o[d] = (_Float16)((xr[d] - mu) * inv * w[d] + b[d]);
}

// ---------------------------------------------------------------------------
// Weight transpose+convert: W[K,N] f32 -> Wt[N,K] f16 (so GEMM B-tiles are
// N-major raw-copyable 16B chunks with contiguous K for the WMMA B fragment)
// ---------------------------------------------------------------------------
__global__ __launch_bounds__(256)
void cvtT_kernel(const float* __restrict__ W, _Float16* __restrict__ Wt,
                 int Kv, int Nv) {
  __shared__ float tle[32][33];
  int n0 = blockIdx.x * 32, k0 = blockIdx.y * 32;
  int tid = threadIdx.x;
#pragma unroll
  for (int e = 0; e < 4; ++e) {
    int p = tid + 256 * e;
    int kl = p >> 5, nl = p & 31;
    tle[kl][nl] = W[(size_t)(k0 + kl) * Nv + n0 + nl];
  }
  __syncthreads();
#pragma unroll
  for (int e = 0; e < 4; ++e) {
    int p = tid + 256 * e;
    int nl = p >> 5, kl = p & 31;
    Wt[(size_t)(n0 + nl) * Kv + k0 + kl] = (_Float16)tle[kl][nl];
  }
}

// ---------------------------------------------------------------------------
// Tiled WMMA GEMM: out = op(Ah[M,K] @ Wt[N,K]^T + bias) (+ resid)
// f16 operands staged via async global->LDS b128; f32 accumulate.
// Block 256 thr = 8 wave32, tile 128x64, wave = 2x2 subtiles of 16x16, K-step 32
// ---------------------------------------------------------------------------
template<bool GELU>
__global__ __launch_bounds__(256)
void gemm_wmma(const _Float16* __restrict__ Ah, const _Float16* __restrict__ Wt,
               const float* __restrict__ bias, const float* __restrict__ resid,
               float* __restrict__ out, _Float16* __restrict__ outh,
               int Mv, int Nv, int Kv) {
  __shared__ _Float16 As[128][40];   // [m][k], row stride 80B (16B aligned)
  __shared__ _Float16 Bs[64][40];    // [n][k]
  int tid  = threadIdx.x;
  int lane = tid & 31;
  int wv   = tid >> 5;
  int wm   = wv & 3;                 // 4 waves along M
  int wn   = wv >> 2;                // 2 waves along N
  int hv   = lane >> 4;              // half-wave id
  int r    = lane & 15;
  int m0 = blockIdx.y * 128;
  int n0 = blockIdx.x * 64;

  v8f acc[2][2] = {};

  for (int k0 = 0; k0 < Kv; k0 += 32) {
    __syncthreads();
    // A tile 128x32 halfs = 512 x 16B chunks (2 per thread)
#pragma unroll
    for (int c = 0; c < 2; ++c) {
      int p = tid + 256 * c;
      int row = p >> 2, col = (p & 3) * 8;
      ASYNC_B128(Ah + (size_t)(m0 + row) * Kv + k0 + col, &As[row][col]);
    }
    // B tile 64x32 halfs = 256 x 16B chunks (1 per thread)
    {
      int n = tid >> 2, col = (tid & 3) * 8;
      ASYNC_B128(Wt + (size_t)(n0 + n) * Kv + k0 + col, &Bs[n][col]);
    }
    WAIT_ASYNC();
    __syncthreads();

    v16h a[2], bf[2];
    int kb = hv * 8;
#pragma unroll
    for (int ti = 0; ti < 2; ++ti) {          // A frags (16x32): ISA layout
      int row = wm * 32 + ti * 16 + r;
#pragma unroll
      for (int e = 0; e < 8; ++e) {
        a[ti][e]     = As[row][kb + e];
        a[ti][8 + e] = As[row][kb + 16 + e];
      }
    }
#pragma unroll
    for (int tj = 0; tj < 2; ++tj) {          // B frags (32x16): ISA layout
      int n = wn * 32 + tj * 16 + r;
#pragma unroll
      for (int e = 0; e < 16; ++e) bf[tj][e] = Bs[n][(e & 8) * 2 + hv * 8 + (e & 7)];
    }
#pragma unroll
    for (int ti = 0; ti < 2; ++ti)
#pragma unroll
      for (int tj = 0; tj < 2; ++tj)
        acc[ti][tj] = __builtin_amdgcn_wmma_f32_16x16x32_f16(
            false, a[ti], false, bf[tj], (short)0, acc[ti][tj], false, false);
  }

#pragma unroll
  for (int ti = 0; ti < 2; ++ti)
#pragma unroll
    for (int tj = 0; tj < 2; ++tj) {
      int mb = m0 + wm * 32 + ti * 16;
      int nb = n0 + wn * 32 + tj * 16;
      float bv = bias ? bias[nb + r] : 0.f;
#pragma unroll
      for (int e = 0; e < 8; ++e) {
        int row = mb + hv * 8 + e;            // C layout: lanes>=16 own M=8+e
        size_t o = (size_t)row * Nv + nb + r;
        float v = acc[ti][tj][e] + bv;
        if (GELU) v = 0.5f * v * (1.f + erff(v * 0.70710678118f));
        if (resid) v += resid[o];
        if (out)  out[o]  = v;
        if (outh) outh[o] = (_Float16)v;
      }
    }
}

// ---------------------------------------------------------------------------
// Flash attention on f16 Q/K/V: one block per (b*h, 128 q-rows), wave = 16 rows.
// K tile raw async-copied to LDS; V tile staged transposed; output f16.
// Q pre-scaled by 1/sqrt(HD)=0.125 (exact in f16).
// ---------------------------------------------------------------------------
__global__ __launch_bounds__(256)
void attn_kernel(const _Float16* __restrict__ qh, const _Float16* __restrict__ kh,
                 const _Float16* __restrict__ vh, _Float16* __restrict__ yh) {
  __shared__ _Float16 Kt[32][72];        // [key][hd]   raw copy of K tile
  __shared__ _Float16 Vt[64][40];        // [hd][key]   transposed V tile
  __shared__ _Float16 Pt[8][16][40];     // per-wave P scratch [qrow][key]
  int tid  = threadIdx.x;
  int lane = tid & 31;
  int wv   = tid >> 5;
  int hv   = lane >> 4;
  int r    = lane & 15;
  int bh = blockIdx.x;
  int b  = bh >> 4;                      // NHEAD = 16
  int h  = bh & 15;
  int qbase = blockIdx.y * 128 + wv * 16;
  int kb8 = hv * 8;

  // Q fragments (A 16x32 layout), 16B vector loads, scaled by 0.125
  v16h aq0, aq1;
  {
    size_t g = ((size_t)(b * TSEQ + qbase + r)) * DIM + h * HDIM + kb8;
    const _Float16 c = (_Float16)0.125f;
    h8 s0 = *(const h8*)(qh + g);
    h8 s1 = *(const h8*)(qh + g + 16);
    h8 s2 = *(const h8*)(qh + g + 32);
    h8 s3 = *(const h8*)(qh + g + 48);
#pragma unroll
    for (int e = 0; e < 8; ++e) {
      aq0[e]     = s0[e] * c;
      aq0[8 + e] = s1[e] * c;
      aq1[e]     = s2[e] * c;
      aq1[8 + e] = s3[e] * c;
    }
  }

  v8f o0 = {}, o1 = {}, o2 = {}, o3 = {};
  float m8[8], l8[8];
#pragma unroll
  for (int e = 0; e < 8; ++e) { m8[e] = -1e30f; l8[e] = 0.f; }

  for (int kb0 = 0; kb0 < TSEQ; kb0 += 32) {
    __syncthreads();
    // K tile: 32 keys x 64 hd halfs = 256 x 16B -> one async copy per thread
    {
      int key = tid >> 3, col = (tid & 7) * 8;
      ASYNC_B128(kh + ((size_t)(b * TSEQ + kb0 + key)) * DIM + h * HDIM + col,
                 &Kt[key][col]);
    }
    // V tile transposed: pairs along hd, scatter into [hd][key]
#pragma unroll
    for (int e = 0; e < 4; ++e) {
      int p = tid + 256 * e;
      int key = p >> 5, hd = (p & 31) * 2;
      h2 pr = *(const h2*)(vh + ((size_t)(b * TSEQ + kb0 + key)) * DIM +
                           h * HDIM + hd);
      Vt[hd][key]     = pr.x;
      Vt[hd + 1][key] = pr.y;
    }
    WAIT_ASYNC();
    __syncthreads();

    // scores for 32 keys = two 16x16 subtiles, contracting HD=64 (2 WMMA each)
    v8f s[2];
#pragma unroll
    for (int sub = 0; sub < 2; ++sub) {
      int key = sub * 16 + r;
      v16h bk0, bk1;
#pragma unroll
      for (int e = 0; e < 16; ++e) {
        bk0[e] = Kt[key][(e & 8) * 2 + hv * 8 + (e & 7)];          // hd 0..31
        bk1[e] = Kt[key][32 + (e & 8) * 2 + hv * 8 + (e & 7)];     // hd 32..63
      }
      v8f z = {};
      z = __builtin_amdgcn_wmma_f32_16x16x32_f16(false, aq0, false, bk0,
                                                 (short)0, z, false, false);
      s[sub] = __builtin_amdgcn_wmma_f32_16x16x32_f16(false, aq1, false, bk1,
                                                 (short)0, z, false, false);
    }

    // online softmax update across the 32 new keys
    float mx[8], p0[8], p1[8], rs[8], al[8];
#pragma unroll
    for (int e = 0; e < 8; ++e) mx[e] = fmaxf(s[0][e], s[1][e]);
#pragma unroll
    for (int d = 1; d < 16; d <<= 1)
#pragma unroll
      for (int e = 0; e < 8; ++e) mx[e] = fmaxf(mx[e], __shfl_xor(mx[e], d, 32));
#pragma unroll
    for (int e = 0; e < 8; ++e) {
      float mn = fmaxf(m8[e], mx[e]);
      al[e] = __expf(m8[e] - mn);
      m8[e] = mn;
      p0[e] = __expf(s[0][e] - mn);
      p1[e] = __expf(s[1][e] - mn);
      rs[e] = p0[e] + p1[e];
    }
#pragma unroll
    for (int d = 1; d < 16; d <<= 1)
#pragma unroll
      for (int e = 0; e < 8; ++e) rs[e] += __shfl_xor(rs[e], d, 32);
#pragma unroll
    for (int e = 0; e < 8; ++e) {
      l8[e] = l8[e] * al[e] + rs[e];
      o0[e] *= al[e]; o1[e] *= al[e]; o2[e] *= al[e]; o3[e] *= al[e];
    }

    // transpose P (C layout -> A layout) through per-wave LDS scratch
#pragma unroll
    for (int e = 0; e < 8; ++e) {
      Pt[wv][hv * 8 + e][r]      = (_Float16)p0[e];
      Pt[wv][hv * 8 + e][16 + r] = (_Float16)p1[e];
    }
    __syncthreads();
    v16h pa;
#pragma unroll
    for (int e = 0; e < 8; ++e) {
      pa[e]     = Pt[wv][r][hv * 16 + e];
      pa[8 + e] = Pt[wv][r][hv * 16 + 8 + e];
    }
    // O += P(16x32) @ V(32x64): 4 output subtiles along hd
    v16h bv;
#pragma unroll
    for (int e = 0; e < 16; ++e) bv[e] = Vt[0  + r][hv * 16 + e];
    o0 = __builtin_amdgcn_wmma_f32_16x16x32_f16(false, pa, false, bv, (short)0, o0, false, false);
#pragma unroll
    for (int e = 0; e < 16; ++e) bv[e] = Vt[16 + r][hv * 16 + e];
    o1 = __builtin_amdgcn_wmma_f32_16x16x32_f16(false, pa, false, bv, (short)0, o1, false, false);
#pragma unroll
    for (int e = 0; e < 16; ++e) bv[e] = Vt[32 + r][hv * 16 + e];
    o2 = __builtin_amdgcn_wmma_f32_16x16x32_f16(false, pa, false, bv, (short)0, o2, false, false);
#pragma unroll
    for (int e = 0; e < 16; ++e) bv[e] = Vt[48 + r][hv * 16 + e];
    o3 = __builtin_amdgcn_wmma_f32_16x16x32_f16(false, pa, false, bv, (short)0, o3, false, false);
  }

  // normalize & store (f16, feeds projection GEMM)
#pragma unroll
  for (int e = 0; e < 8; ++e) {
    float inv = l8[e] > 0.f ? 1.f / l8[e] : 0.f;
    size_t row = (size_t)(b * TSEQ + qbase + hv * 8 + e);
    size_t base = row * DIM + h * HDIM + r;
    yh[base +  0] = (_Float16)(o0[e] * inv);
    yh[base + 16] = (_Float16)(o1[e] * inv);
    yh[base + 32] = (_Float16)(o2[e] * inv);
    yh[base + 48] = (_Float16)(o3[e] * inv);
  }
}

// ---------------------------------------------------------------------------
extern "C" void kernel_launch(void* const* d_in, const int* in_sizes, int n_in,
                              void* d_out, int out_size, void* d_ws, size_t ws_size,
                              hipStream_t stream) {
  const int*   idx   = (const int*)  d_in[0];
  const int*   st    = (const int*)  d_in[1];
  const int*   tt    = (const int*)  d_in[2];
  const float* tok   = (const float*)d_in[3];
  const float* pos   = (const float*)d_in[4];
  const float* sem   = (const float*)d_in[5];
  const float* tex   = (const float*)d_in[6];
  const float* ln1w  = (const float*)d_in[7];
  const float* ln1b  = (const float*)d_in[8];
  const float* Wq    = (const float*)d_in[9];
  const float* bq    = (const float*)d_in[10];
  const float* Wk    = (const float*)d_in[11];
  const float* bk    = (const float*)d_in[12];
  const float* Wv    = (const float*)d_in[13];
  const float* bv    = (const float*)d_in[14];
  const float* Wp    = (const float*)d_in[15];
  const float* bp    = (const float*)d_in[16];
  const float* ln2w  = (const float*)d_in[17];
  const float* ln2b  = (const float*)d_in[18];
  const float* W1    = (const float*)d_in[19];
  const float* b1    = (const float*)d_in[20];
  const float* W2    = (const float*)d_in[21];
  const float* b2    = (const float*)d_in[22];
  const float* lnfw  = (const float*)d_in[23];
  const float* lnfb  = (const float*)d_in[24];
  const float* headw = (const float*)d_in[25];
  float* out = (float*)d_out;

  // workspace (64 MB):
  //   x (f32 residual stream) | hh | qh kh vh yh (f16; span reused as MLP
  //   hidden ROWSxFF) | Wth (f16 transposed weight staging, max DIM*FF)
  float*    x    = (float*)d_ws;
  _Float16* hh   = (_Float16*)(x + (size_t)ROWS * DIM);
  _Float16* qbh  = hh  + (size_t)ROWS * DIM;
  _Float16* kbh  = qbh + (size_t)ROWS * DIM;
  _Float16* vbh  = kbh + (size_t)ROWS * DIM;
  _Float16* ybh  = vbh + (size_t)ROWS * DIM;
  _Float16* hidh = qbh;                         // ROWS*FF halfs spanning q..y
  _Float16* Wth  = ybh + (size_t)ROWS * DIM;    // DIM*FF halfs

  dim3 blk(256);
  dim3 gLN(ROWS);
  dim3 gDD(DIM / 64, ROWS / 128);      // N=1024 GEMMs
  dim3 gFF(FF / 64, ROWS / 128);       // N=4096 GEMM
  dim3 gHD(CLS / 64, ROWS / 128);      // N=256 head GEMMs
  dim3 gAT(BATCH * NHEAD, TSEQ / 128); // attention
  dim3 gTdd(DIM / 32, DIM / 32);       // W transpose grids
  dim3 gTdf(FF / 32, DIM / 32);        // W1: K=DIM,N=FF
  dim3 gTfd(DIM / 32, FF / 32);        // W2: K=FF,N=DIM
  dim3 gThd(CLS / 32, DIM / 32);       // head: K=DIM,N=CLS

  embed_kernel<<<gLN, blk, 0, stream>>>(idx, st, tt, tok, pos, sem, tex, x);

  for (int l = 0; l < LAY; ++l) {
    size_t oD  = (size_t)l * DIM;
    size_t oDD = (size_t)l * DIM * DIM;
    size_t oF  = (size_t)l * FF;
    size_t oDF = (size_t)l * DIM * FF;

    ln_kernel<<<gLN, blk, 0, stream>>>(x, ln1w + oD, ln1b + oD, hh);
    cvtT_kernel<<<gTdd, blk, 0, stream>>>(Wq + oDD, Wth, DIM, DIM);
    gemm_wmma<false><<<gDD, blk, 0, stream>>>(hh, Wth, bq + oD, nullptr, nullptr, qbh, ROWS, DIM, DIM);
    cvtT_kernel<<<gTdd, blk, 0, stream>>>(Wk + oDD, Wth, DIM, DIM);
    gemm_wmma<false><<<gDD, blk, 0, stream>>>(hh, Wth, bk + oD, nullptr, nullptr, kbh, ROWS, DIM, DIM);
    cvtT_kernel<<<gTdd, blk, 0, stream>>>(Wv + oDD, Wth, DIM, DIM);
    gemm_wmma<false><<<gDD, blk, 0, stream>>>(hh, Wth, bv + oD, nullptr, nullptr, vbh, ROWS, DIM, DIM);

    attn_kernel<<<gAT, blk, 0, stream>>>(qbh, kbh, vbh, ybh);

    cvtT_kernel<<<gTdd, blk, 0, stream>>>(Wp + oDD, Wth, DIM, DIM);
    gemm_wmma<false><<<gDD, blk, 0, stream>>>(ybh, Wth, bp + oD, x, x, nullptr, ROWS, DIM, DIM);

    ln_kernel<<<gLN, blk, 0, stream>>>(x, ln2w + oD, ln2b + oD, hh);
    cvtT_kernel<<<gTdf, blk, 0, stream>>>(W1 + oDF, Wth, DIM, FF);
    gemm_wmma<true ><<<gFF, blk, 0, stream>>>(hh, Wth, b1 + oF, nullptr, nullptr, hidh, ROWS, FF, DIM);
    cvtT_kernel<<<gTfd, blk, 0, stream>>>(W2 + oDF, Wth, FF, DIM);
    gemm_wmma<false><<<gDD, blk, 0, stream>>>(hidh, Wth, b2 + oD, x, x, nullptr, ROWS, DIM, FF);
  }

  ln_kernel<<<gLN, blk, 0, stream>>>(x, lnfw, lnfb, hh);
  for (int kh2 = 0; kh2 < NOUT; ++kh2) {
    cvtT_kernel<<<gThd, blk, 0, stream>>>(headw + (size_t)kh2 * DIM * CLS, Wth, DIM, CLS);
    gemm_wmma<false><<<gHD, blk, 0, stream>>>(
        hh, Wth, nullptr, nullptr, out + (size_t)kh2 * ROWS * CLS, nullptr,
        ROWS, CLS, DIM);
  }
}
